// TransformerBlockQuantum_65481071404323
// MI455X (gfx1250) — matrix operand order repaired and checked by reading, not compiled, not run on hardware
//
#include <hip/hip_runtime.h>
#include <hip/hip_bf16.h>

#define B_   32
#define S_   1024
#define E_   8
#define H_   2
#define D_   4
#define FFN_ 32
#define EPS_ 1e-5f

typedef float    v2f  __attribute__((ext_vector_type(2)));
typedef float    v8f  __attribute__((ext_vector_type(8)));
typedef _Float16 v8h  __attribute__((ext_vector_type(8)));
typedef _Float16 v16h __attribute__((ext_vector_type(16)));
typedef unsigned int u32x4 __attribute__((ext_vector_type(4)));
typedef int          i32x4 __attribute__((ext_vector_type(4)));
typedef int          i32x8 __attribute__((ext_vector_type(8)));

// ---------------------------------------------------------------------------
// TDM: 1-D contiguous tensor load (global -> LDS) via Tensor Data Mover.
// D# built per CDNA5 ISA 8.3/8.4: group0 = {flags, lds_addr, global_addr,
// type=2}, group1 = {data_size, tensor_dim0/1, tile_dim0/1/2, strides}.
// nelem <= 65535, no padding/iteration/multicast (workgroup_mask = 0).
// dsize_code: 0=1B, 1=2B, 2=4B, 3=8B.
// clang-23 toolchain: 6-arg builtin (extra int32x8 operand, zero-filled).
// ---------------------------------------------------------------------------
__device__ __forceinline__ void tdm_load_1d(unsigned int lds_addr,
                                            const void* gptr,
                                            unsigned int nelem,
                                            int dsize_code) {
  const unsigned long long ga = (unsigned long long)gptr;
  u32x4 g0;
  g0[0] = 1u;                                        // count=1, user desc
  g0[1] = lds_addr;                                  // lds_addr [63:32]
  g0[2] = (unsigned int)ga;                          // global_addr lo
  g0[3] = (unsigned int)((ga >> 32) & 0x1FFFFFFu)    // global_addr [56:32]
        | (2u << 30);                                // type=2 ("image")
  i32x8 g1;
  g1[0] = dsize_code << 16;                          // data_size; mask/flags=0
  g1[1] = (int)(nelem << 16);                        // tensor_dim0 lo16 @ [63:48]
  g1[2] = (int)(1u << 16);                           // dim0 hi16=0; tensor_dim1=1
  g1[3] = (int)(nelem << 16);                        // dim1 hi16=0; tile_dim0
  g1[4] = 1;                                         // tile_dim1=1; tile_dim2=0
  g1[5] = (int)nelem;                                // tensor_dim0_stride lo32
  g1[6] = (int)(nelem << 16);                        // stride0 hi=0; stride1 lo16
  g1[7] = 0;                                         // stride1 hi
  i32x4 z4 = {0, 0, 0, 0};                           // groups 2/3 unused (2-D)
  i32x8 z8 = {0, 0, 0, 0, 0, 0, 0, 0};
  __builtin_amdgcn_tensor_load_to_lds(g0, g1, z4, z4, z8, 0);
}

// ---------------------------------------------------------------------------
// Kernel 1: QKV projection.  One thread per token. q pre-scaled by 1/sqrt(D).
// q,k -> [B,H,S,D] f32;  v -> [B,H,D,S] f16 (pre-transposed for the PV WMMA
// B-operand so the attention kernel can TDM it straight into LDS).
// ---------------------------------------------------------------------------
__global__ __launch_bounds__(256) void qkv_kernel(
    const float* __restrict__ x, const float* __restrict__ wqkv,
    const float* __restrict__ bqkv,
    float* __restrict__ q, float* __restrict__ k, _Float16* __restrict__ vt) {
  __shared__ float sw[24 * 8];
  __shared__ float sb[24];
  const int tid = threadIdx.x;
  if (tid < 192) sw[tid] = wqkv[tid];
  if (tid < 24)  sb[tid] = bqkv[tid];
  __syncthreads();

  const int t = blockIdx.x * 256 + tid;        // token id, 0..32767
  const int b = t >> 10;
  const int s = t & 1023;
  float xv[8];
#pragma unroll
  for (int e = 0; e < 8; ++e) xv[e] = x[(size_t)t * 8 + e];

#pragma unroll
  for (int o = 0; o < 24; ++o) {
    float acc = sb[o];
#pragma unroll
    for (int e = 0; e < 8; ++e) acc += sw[o * 8 + e] * xv[e];
    const int h = (o >> 2) & 1;
    const int d = o & 3;
    if (o < 8) {
      q[(((size_t)b * H_ + h) * S_ + s) * D_ + d] = acc * 0.5f;
    } else if (o < 16) {
      k[(((size_t)b * H_ + h) * S_ + s) * D_ + d] = acc;
    } else {
      vt[(((size_t)b * H_ + h) * D_ + d) * S_ + s] = (_Float16)acc;
    }
  }
}

// ---------------------------------------------------------------------------
// Kernel 2: attention.  Block = (b*H+h, slab).  8 waves, each owns 2 query
// tiles of 16.  K (f32) and V^T (f16) pulled into LDS by the Tensor Data
// Mover (TENSORcnt-tracked), issued once by wave 0.
// Scores: V_WMMA_F32_16X16X4_F32 (K=4 == head dim, no padding).
// P@V:    V_WMMA_F32_16X16X32_F16, P transposed through per-wave LDS staging.
// ---------------------------------------------------------------------------
__global__ __launch_bounds__(256) void attn_kernel(
    const float* __restrict__ q, const float* __restrict__ k,
    const _Float16* __restrict__ vt, float* __restrict__ o) {
  __shared__ float    sK[S_ * D_];        // 16 KB, [key][dim] f32
  __shared__ _Float16 sVt[D_ * S_];       //  8 KB, [dim][key] f16
  __shared__ _Float16 sP[8][16 * 32];     //  8 KB, per-wave P staging

  const int bh   = blockIdx.x;            // 0..63  (b*H + h)
  const int slab = blockIdx.y;            // 0..3
  const int tid  = threadIdx.x;
  const int lane = tid & 31;
  const int w    = tid >> 5;
  const int hi   = (lane >= 16) ? 1 : 0;
  const int l16  = lane & 15;

  // Wave 0 DMAs K and V^T for this (b,h) into LDS; EXEC is ignored by TDM.
  if (w == 0) {
    const unsigned int ldsK = (unsigned int)(unsigned long long)(void*)&sK[0];
    const unsigned int ldsV = (unsigned int)(unsigned long long)(void*)&sVt[0];
    tdm_load_1d(ldsK, k  + (size_t)bh * S_ * D_, S_ * D_, /*4B*/ 2);
    tdm_load_1d(ldsV, vt + (size_t)bh * S_ * D_, S_ * D_, /*2B*/ 1);
    __builtin_amdgcn_s_wait_tensorcnt(0);
  }
  __syncthreads();

  const int b = bh >> 1, h = bh & 1;

  for (int t = 0; t < 2; ++t) {
    const int qt = slab * 16 + t * 8 + w;   // query tile 0..63
    const int q0 = qt * 16;

    // Q A-fragment (16x4 f32): lanes 0-15 -> K=0,1 ; lanes 16-31 -> K=2,3
    const v2f aq = *(const v2f*)(q + ((size_t)bh * S_ + q0 + l16) * D_ + hi * 2);

    v8f oacc = {};
    float rs[8];
#pragma unroll
    for (int r = 0; r < 8; ++r) rs[r] = 0.0f;

    for (int c = 0; c < S_ / 32; ++c) {       // 32-key chunks
#pragma unroll
      for (int half = 0; half < 2; ++half) {
        const int k0 = c * 32 + half * 16;
        // B-fragment = K^T (4x16 f32), column n = l16 -> key k0+n
        const v2f bk = *(const v2f*)(&sK[(k0 + l16) * D_ + hi * 2]);
        v8f cz = {};
        v8f d = __builtin_amdgcn_wmma_f32_16x16x4_f32(
            false, aq, false, bk, (short)0, cz, false, false);
        // exp + rowsum + stage P as f16 (row-major [16 q][32 k])
        const int koff = half * 16 + l16;
#pragma unroll
        for (int r = 0; r < 8; ++r) {
          const float p = __expf(d[r]);
          rs[r] += p;
          sP[w][(r + hi * 8) * 32 + koff] = (_Float16)p;
        }
      }
      // A-fragment of P (16x32 f16): lanes0-15 M=l16, K=0-7/16-23;
      //                              lanes16-31 M=l16, K=8-15/24-31
      union { v16h vv; v8h h2[2]; } pa, vb;
      {
        const _Float16* base = &sP[w][l16 * 32 + hi * 8];
        pa.h2[0] = *(const v8h*)(base);
        pa.h2[1] = *(const v8h*)(base + 16);
        // B-fragment of V (32x16 f16): column n = l16 (dims 0..3 valid),
        // lanes0-15 K=c*32+0-7/16-23, lanes16-31 K=c*32+8-15/24-31
        const int nc = l16 & 3;                 // clamp unused cols to valid data
        const _Float16* vtb = &sVt[nc * S_ + c * 32 + hi * 8];
        vb.h2[0] = *(const v8h*)(vtb);
        vb.h2[1] = *(const v8h*)(vtb + 16);
      }
      oacc = __builtin_amdgcn_wmma_f32_16x16x32_f16(
          false, pa.vv, false, vb.vv, (short)0, oacc, false, false);
    }

    // reduce rowsums across the 16-lane half (rows live in VGPR index)
#pragma unroll
    for (int r = 0; r < 8; ++r) {
      float s = rs[r];
      s += __shfl_xor(s, 1, 32);
      s += __shfl_xor(s, 2, 32);
      s += __shfl_xor(s, 4, 32);
      s += __shfl_xor(s, 8, 32);
      rs[r] = s;
    }
    // store O: lane column = l16 (only 0..3 valid), rows r + 8*hi
    if (l16 < 4) {
#pragma unroll
      for (int r = 0; r < 8; ++r) {
        const int row = r + hi * 8;
        o[(((size_t)b * S_ + q0 + row) * H_ + h) * D_ + l16] = oacc[r] / rs[r];
      }
    }
  }
}

// ---------------------------------------------------------------------------
// Kernel 3: out-proj + residual + LN1 + quantum cos + FFN + residual + LN2.
// One thread per token; weights staged in LDS.
// ---------------------------------------------------------------------------
__global__ __launch_bounds__(256) void epilogue_kernel(
    const float* __restrict__ x,    const float* __restrict__ oattn,
    const float* __restrict__ wout, const float* __restrict__ bout,
    const float* __restrict__ g1,   const float* __restrict__ be1,
    const float* __restrict__ th,
    const float* __restrict__ w1,   const float* __restrict__ bf1,
    const float* __restrict__ w2,   const float* __restrict__ bf2,
    const float* __restrict__ g2,   const float* __restrict__ be2,
    float* __restrict__ out) {
  __shared__ float s_wout[64], s_w1[256], s_w2[256];
  __shared__ float s_bout[8], s_g1[8], s_be1[8], s_th[8];
  __shared__ float s_bf1[32], s_bf2[8], s_g2[8], s_be2[8];
  const int tid = threadIdx.x;
  if (tid < 64)  s_wout[tid] = wout[tid];
  s_w1[tid] = w1[tid];
  s_w2[tid] = w2[tid];
  if (tid < 32)  s_bf1[tid] = bf1[tid];
  if (tid < 8) {
    s_bout[tid] = bout[tid]; s_g1[tid] = g1[tid]; s_be1[tid] = be1[tid];
    s_th[tid]   = th[tid];   s_bf2[tid] = bf2[tid];
    s_g2[tid]   = g2[tid];   s_be2[tid] = be2[tid];
  }
  __syncthreads();

  const int t = blockIdx.x * 256 + tid;
  float xv[8], ov[8], y[8];
#pragma unroll
  for (int e = 0; e < 8; ++e) {
    xv[e] = x[(size_t)t * 8 + e];
    ov[e] = oattn[(size_t)t * 8 + e];
  }
#pragma unroll
  for (int e = 0; e < 8; ++e) {
    float acc = s_bout[e];
#pragma unroll
    for (int j = 0; j < 8; ++j) acc += s_wout[e * 8 + j] * ov[j];
    y[e] = xv[e] + acc;
  }
  // LN1
  float mu = 0.f;
#pragma unroll
  for (int e = 0; e < 8; ++e) mu += y[e];
  mu *= 0.125f;
  float var = 0.f;
#pragma unroll
  for (int e = 0; e < 8; ++e) { const float d = y[e] - mu; var += d * d; }
  var *= 0.125f;
  const float rstd1 = rsqrtf(var + EPS_);
  float x1[8], qo[8];
#pragma unroll
  for (int e = 0; e < 8; ++e) {
    x1[e] = (y[e] - mu) * rstd1 * s_g1[e] + s_be1[e];
    qo[e] = __cosf(x1[e] + s_th[e]);          // quantum layer closed form
  }
  // FFN
  float hb[32];
#pragma unroll
  for (int f = 0; f < 32; ++f) {
    float acc = s_bf1[f];
#pragma unroll
    for (int e = 0; e < 8; ++e) acc += s_w1[f * 8 + e] * qo[e];
    hb[f] = fmaxf(acc, 0.f);
  }
  float z[8];
#pragma unroll
  for (int e = 0; e < 8; ++e) {
    float acc = s_bf2[e];
#pragma unroll
    for (int f = 0; f < 32; ++f) acc += s_w2[e * 32 + f] * hb[f];
    z[e] = x1[e] + acc;
  }
  // LN2
  float mu2 = 0.f;
#pragma unroll
  for (int e = 0; e < 8; ++e) mu2 += z[e];
  mu2 *= 0.125f;
  float var2 = 0.f;
#pragma unroll
  for (int e = 0; e < 8; ++e) { const float d = z[e] - mu2; var2 += d * d; }
  var2 *= 0.125f;
  const float rstd2 = rsqrtf(var2 + EPS_);
#pragma unroll
  for (int e = 0; e < 8; ++e)
    out[(size_t)t * 8 + e] = (z[e] - mu2) * rstd2 * s_g2[e] + s_be2[e];
}

// ---------------------------------------------------------------------------
extern "C" void kernel_launch(void* const* d_in, const int* in_sizes, int n_in,
                              void* d_out, int out_size, void* d_ws, size_t ws_size,
                              hipStream_t stream) {
  const float* x    = (const float*)d_in[0];
  const float* wqkv = (const float*)d_in[1];
  const float* bqkv = (const float*)d_in[2];
  const float* wout = (const float*)d_in[3];
  const float* bout = (const float*)d_in[4];
  const float* g1   = (const float*)d_in[5];
  const float* be1  = (const float*)d_in[6];
  const float* th   = (const float*)d_in[7];
  const float* w1   = (const float*)d_in[8];
  const float* bf1  = (const float*)d_in[9];
  const float* w2   = (const float*)d_in[10];
  const float* bf2  = (const float*)d_in[11];
  const float* g2   = (const float*)d_in[12];
  const float* be2  = (const float*)d_in[13];
  float* out = (float*)d_out;

  const size_t N = (size_t)B_ * H_ * S_ * D_;   // 262144 elems per tensor
  float*     qws  = (float*)d_ws;
  float*     kws  = qws + N;
  _Float16*  vtws = (_Float16*)(kws + N);       // [B,H,D,S] f16
  float*     ows  = (float*)(vtws + N);         // [B,S,E] attention output

  qkv_kernel<<<dim3((B_ * S_) / 256), dim3(256), 0, stream>>>(
      x, wqkv, bqkv, qws, kws, vtws);
  attn_kernel<<<dim3(B_ * H_, 4), dim3(256), 0, stream>>>(
      qws, kws, vtws, ows);
  epilogue_kernel<<<dim3((B_ * S_) / 256), dim3(256), 0, stream>>>(
      x, ows, wout, bout, g1, be1, th, w1, bf1, w2, bf2, g2, be2, out);
}
